// Decoder_pro_3040836845743
// MI455X (gfx1250) — matrix operand order repaired and checked by reading, not compiled, not run on hardware
//
#include <hip/hip_runtime.h>
#include <math.h>

// ---------------------------------------------------------------------------
// Types for CDNA5 WMMA
// ---------------------------------------------------------------------------
typedef __attribute__((ext_vector_type(16))) _Float16 v16h;
typedef __attribute__((ext_vector_type(8)))  float    v8f;
typedef __attribute__((ext_vector_type(4)))  float    v4f;

#define BN_B 8

// ---------------------------------------------------------------------------
__global__ void zero_f32_kernel(float* __restrict__ p, int n) {
    int i = blockIdx.x * blockDim.x + threadIdx.x;
    if (i < n) p[i] = 0.0f;
}

// ---------------------------------------------------------------------------
// Pack conv weights (f32 OIHW, [Cout][Cin][3][3]) into WMMA A-fragment layout.
// K ordering is TAP-MAJOR: k = t*Cin + ci  (t = 3x3 tap index 0..8).
//   element i = ((tile*chunks + chunk)*32 + lane)*16 + j
//   holds W[m = tile*16 + (lane&15)][k = chunk*32 + kidx(lane,j)]  (f16, 0-pad)
//   kidx = 16*(lane>=16) + j (contiguous k order, matching the B fragment).
// ---------------------------------------------------------------------------
__global__ void pack_wgt_kernel(const float* __restrict__ Wsrc, _Float16* __restrict__ Apack,
                                int Cout, int Cin) {
    const int K      = Cin * 9;
    const int chunks = (K + 31) >> 5;
    const int tiles  = (Cout + 15) >> 4;
    int i = blockIdx.x * blockDim.x + threadIdx.x;
    int total = tiles * chunks * 512;
    if (i >= total) return;
    int j     = i & 15;
    int lane  = (i >> 4) & 31;
    int chunk = (i >> 9) % chunks;
    int tile  = (i >> 9) / chunks;
    int r     = j >> 1;
    int kidx  = ((lane >= 16) ? 16 : 0) + j;
    int k     = chunk * 32 + kidx;
    int m     = tile * 16 + (lane & 15);
    _Float16 v = (_Float16)0.0f;
    if (m < Cout && k < K) {
        int t  = k / Cin;          // tap
        int ci = k - t * Cin;      // input channel
        v = (_Float16)Wsrc[((size_t)m * Cin + ci) * 9 + t];
    }
    Apack[i] = v;
}

// ---------------------------------------------------------------------------
// NCHW f32 -> NHWC f16 (one thread per pixel; reads coalesced across threads)
// ---------------------------------------------------------------------------
__global__ void nchw_to_nhwc_f16_kernel(const float* __restrict__ src, _Float16* __restrict__ dst,
                                        int Bn, int C, int HW) {
    int p = blockIdx.x * blockDim.x + threadIdx.x;
    int total = Bn * HW;
    if (p >= total) return;
    int b = p / HW, hw = p - b * HW;
    _Float16* o = dst + (size_t)p * C;
    for (int c = 0; c < C; ++c)
        o[c] = (_Float16)src[((size_t)b * C + c) * HW + hw];
}

// ---------------------------------------------------------------------------
// 1x1 conv (projection): X f16 NHWC (B,HW,Cin), W f32 (Cmid,Cin), bias f32
// -> Y f32 NCHW (B,Cmid,HW)
// ---------------------------------------------------------------------------
__global__ void conv1x1_kernel(const _Float16* __restrict__ X, const float* __restrict__ Wp,
                               const float* __restrict__ bp, float* __restrict__ Y,
                               int Bn, int Cin, int HW, int Cmid) {
    int i = blockIdx.x * blockDim.x + threadIdx.x;
    int total = Bn * Cmid * HW;
    if (i >= total) return;
    int hw = i % HW;
    int t  = i / HW;
    int cc = t % Cmid;
    int b  = t / Cmid;
    float s = bp[cc];
    const _Float16* xb = X + ((size_t)b * HW + hw) * Cin;   // contiguous per thread
    const float*    wr = Wp + (size_t)cc * Cin;
    for (int ci = 0; ci < Cin; ++ci)
        s += wr[ci] * (float)xb[ci];
    Y[i] = s;
}

// ---------------------------------------------------------------------------
// 2x bilinear (align-corners) upsample of Pj (NCHW f32, H=H2/2) + concat with
// Xs (NCHW f32) -> Out NHWC f16 (B, H2*W2, Cs+Cm). One thread per pixel.
// ---------------------------------------------------------------------------
__global__ void upcat_nhwc_kernel(const float* __restrict__ Xs, const float* __restrict__ Pj,
                                  _Float16* __restrict__ Out,
                                  int Bn, int Cs, int Cm, int H2, int W2) {
    const int H = H2 >> 1, W = W2 >> 1;
    int p = blockIdx.x * blockDim.x + threadIdx.x;
    int total = Bn * H2 * W2;
    if (p >= total) return;
    int x = p % W2;
    int t = p / W2;
    int y = t % H2;
    int b = t / H2;
    _Float16* o = Out + (size_t)p * (Cs + Cm);
    for (int c = 0; c < Cs; ++c)
        o[c] = (_Float16)Xs[(((size_t)b * Cs + c) * H2 + y) * W2 + x];
    float fy = (float)y * ((float)(H - 1) / (float)(H2 - 1));
    float fx = (float)x * ((float)(W - 1) / (float)(W2 - 1));
    int y0 = (int)floorf(fy); int y1 = min(y0 + 1, H - 1); float wy = fy - (float)y0;
    int x0 = (int)floorf(fx); int x1 = min(x0 + 1, W - 1); float wx = fx - (float)x0;
    for (int cc = 0; cc < Cm; ++cc) {
        const float* pp = Pj + ((size_t)b * Cm + cc) * H * W;
        float r0 = pp[y0 * W + x0] * (1.f - wy) + pp[y1 * W + x0] * wy;
        float r1 = pp[y0 * W + x1] * (1.f - wy) + pp[y1 * W + x1] * wy;
        o[Cs + cc] = (_Float16)(r0 * (1.f - wx) + r1 * wx);
    }
}

// ---------------------------------------------------------------------------
// 3x3 SAME conv, implicit GEMM on WMMA f16->f32, NHWC activations.
//   M = Cout (MT 16-row tiles/wave), N = B*H*W, K = Cin*9 (tap-major).
//   blockDim = 128 (4 waves, each an independent 16-col N tile).
//   A fragments pre-packed; both A and B fragments load as whole v16h
//   (2 x global_load_b128 each) — no per-element gathering.
//   Every 16-k lane window lies in one tap (Cin multiple of 16), so the
//   B fragment is 16 contiguous halves of one (possibly shifted) pixel,
//   zeroed branchlessly when out of bounds.
// OUT_NCHW=true (final conv) stores strided NCHW with m<Cout guard;
// otherwise stores 2 x float4 contiguous NHWC.
// ---------------------------------------------------------------------------
template<int MT, bool OUT_NCHW>
__global__ void conv3x3_wmma_kernel(const _Float16* __restrict__ X,
                                    const _Float16* __restrict__ Apack,
                                    const float* __restrict__ bias,
                                    float* __restrict__ Y,
                                    int Bn, int Cin, int cinLog2,
                                    int H, int Wd, int Cout) {
    const int K      = Cin * 9;
    const int chunks = (K + 31) >> 5;
    const int HW     = H * Wd;
    const int Npix   = Bn * HW;

    const int wave = threadIdx.x >> 5;
    const int lane = threadIdx.x & 31;
    const int nTileBase = (blockIdx.x * (blockDim.x >> 5) + wave) * 16;
    const int tile0     = blockIdx.y * MT;
    if (nTileBase >= Npix) return;  // wave-uniform: EXEC stays all-ones for WMMA

    const int nB     = nTileBase + (lane & 15);
    const int validN = (nB < Npix) ? 1 : 0;
    const int nC     = validN ? nB : (Npix - 1);
    const int b      = nC / HW;
    const int hw     = nC - b * HW;
    const int h      = hw / Wd;
    const int w      = hw - h * Wd;

    const int kB_base = (lane >= 16) ? 16 : 0;
    const int mHalf   = (lane >= 16) ? 8 : 0;

    v8f acc[MT];
#pragma unroll
    for (int mt = 0; mt < MT; ++mt)
#pragma unroll
        for (int r = 0; r < 8; ++r) {
            int m = (tile0 + mt) * 16 + r + mHalf;
            acc[mt][r] = (m < Cout) ? bias[m] : 0.0f;
        }

    const _Float16* aBase = Apack + ((size_t)tile0 * chunks * 32 + lane) * 16;
    const size_t aTileStride = (size_t)chunks * 512;   // halves per 16-row tile
    const _Float16* xBat = X + (size_t)b * HW * Cin;   // NHWC batch base

    for (int k0 = 0; k0 < K; k0 += 32) {
        const int chunk = k0 >> 5;
        // ---- B fragment: 16 contiguous halves of one shifted pixel --------
        const int kb  = k0 + kB_base;                  // start of this lane's window
        const int t   = kb >> cinLog2;                 // tap (may be 9 when padded)
        const int ciB = kb - (t << cinLog2);
        const int tc  = min(t, 8);
        const int r3  = tc / 3;
        const int ih  = h + r3 - 1;
        const int iw  = w + (tc - r3 * 3) - 1;
        const int ok  = validN & (int)(kb < K) & (int)(ih >= 0) & (int)(ih < H)
                               & (int)(iw >= 0) & (int)(iw < Wd);
        const int ihc = min(max(ih, 0), H - 1);
        const int iwc = min(max(iw, 0), Wd - 1);
        const v16h zz = {};
        v16h bb = *(const v16h*)(xBat + ((size_t)ihc * Wd + iwc) * Cin + ciB);
        bb = ok ? bb : zz;
        // ---- MT WMMAs sharing this B fragment -----------------------------
#pragma unroll
        for (int mt = 0; mt < MT; ++mt) {
            const v16h a = *(const v16h*)(aBase + (size_t)mt * aTileStride + (size_t)chunk * 512);
            acc[mt] = __builtin_amdgcn_wmma_f32_16x16x32_f16(false, a, false, bb,
                                                             (short)0, acc[mt], false, false);
        }
    }

    if (validN) {
        if (OUT_NCHW) {
            const size_t outBase = (size_t)b * Cout * HW + hw;
#pragma unroll
            for (int mt = 0; mt < MT; ++mt)
#pragma unroll
                for (int r = 0; r < 8; ++r) {
                    int m = (tile0 + mt) * 16 + r + mHalf;
                    if (m < Cout) Y[outBase + (size_t)m * HW] = acc[mt][r];
                }
        } else {
            float* o = Y + (size_t)nC * Cout;
#pragma unroll
            for (int mt = 0; mt < MT; ++mt) {
                const int m0 = (tile0 + mt) * 16 + mHalf;
                v4f lo = {acc[mt][0], acc[mt][1], acc[mt][2], acc[mt][3]};
                v4f hi = {acc[mt][4], acc[mt][5], acc[mt][6], acc[mt][7]};
                *(v4f*)(o + m0)     = lo;
                *(v4f*)(o + m0 + 4) = hi;
            }
        }
    }
}

// ---------------------------------------------------------------------------
// BN stats on NHWC raw: cooperative coalesced partial sums -> atomics.
// sums[0..128): sum per channel; sums[128..256): sum of squares.
// ---------------------------------------------------------------------------
__global__ void bn_partial_kernel(const float* __restrict__ Y, float* __restrict__ sums,
                                  int C, int Npx) {
    const int tpc = blockDim.x / C;        // pixels covered per block iteration
    const int c   = threadIdx.x % C;
    const int sub = threadIdx.x / C;
    float s = 0.f, s2 = 0.f;
    const int stride = gridDim.x * tpc;
    for (int p = blockIdx.x * tpc + sub; p < Npx; p += stride) {
        float v = Y[(size_t)p * C + c];
        s += v; s2 += v * v;
    }
    __shared__ float sh1[256], sh2[256];
    sh1[threadIdx.x] = s; sh2[threadIdx.x] = s2;
    __syncthreads();
    for (int off = blockDim.x >> 1; off >= C; off >>= 1) {
        if ((int)threadIdx.x < off) {
            sh1[threadIdx.x] += sh1[threadIdx.x + off];
            sh2[threadIdx.x] += sh2[threadIdx.x + off];
        }
        __syncthreads();
    }
    if ((int)threadIdx.x < C) {
        atomicAdd(&sums[c],       sh1[threadIdx.x]);
        atomicAdd(&sums[128 + c], sh2[threadIdx.x]);
    }
}

__global__ void bn_finalize_kernel(const float* __restrict__ sums, float* __restrict__ mv,
                                   float* __restrict__ iv, int C, float n) {
    int c = blockIdx.x * blockDim.x + threadIdx.x;
    if (c >= C) return;
    float m = sums[c] / n;
    float var = sums[128 + c] / n - m * m;
    mv[c] = m;
    iv[c] = rsqrtf(var + 1e-5f);
}

// ---------------------------------------------------------------------------
// BN apply + LeakyReLU on NHWC; writes f16 activation (and optional f32 copy)
// ---------------------------------------------------------------------------
__global__ void bn_lrelu_kernel(const float* __restrict__ Y, const float* __restrict__ mv,
                                const float* __restrict__ iv, const float* __restrict__ g,
                                const float* __restrict__ beta,
                                _Float16* __restrict__ actH, float* __restrict__ actF,
                                int C, int total) {
    int i = blockIdx.x * blockDim.x + threadIdx.x;
    if (i >= total) return;
    int c = i % C;     // C is a power of two
    float v = (Y[i] - mv[c]) * iv[c] * g[c] + beta[c];
    v = (v >= 0.f) ? v : 0.01f * v;
    actH[i] = (_Float16)v;
    if (actF) actF[i] = v;
}

// ---------------------------------------------------------------------------
// Prototype accumulation: softmax(logits NCHW) over 4 classes; xf NHWC.
// acc[b][k][0..15] = sum_hw mask*x ; acc[b][k][16] = sum_hw mask
// ---------------------------------------------------------------------------
__global__ void proto_accum_kernel(const float* __restrict__ logits, const float* __restrict__ xf,
                                   float* __restrict__ acc, int HW) {
    const int b = blockIdx.x;
    float a[4][16];
    float ms[4];
#pragma unroll
    for (int k = 0; k < 4; ++k) {
        ms[k] = 0.f;
#pragma unroll
        for (int c = 0; c < 16; ++c) a[k][c] = 0.f;
    }
    for (int p = threadIdx.x; p < HW; p += blockDim.x) {
        float l[4]; float mx = -1e30f;
#pragma unroll
        for (int k = 0; k < 4; ++k) {
            l[k] = logits[((size_t)(b * 4 + k)) * HW + p];
            mx = fmaxf(mx, l[k]);
        }
        float e[4]; float se = 0.f;
#pragma unroll
        for (int k = 0; k < 4; ++k) { e[k] = expf(l[k] - mx); se += e[k]; }
        float inv = 1.f / se;
        const float* xv = xf + ((size_t)b * HW + p) * 16;   // contiguous feature
#pragma unroll
        for (int k = 0; k < 4; ++k) {
            float m = e[k] * inv;
            ms[k] += m;
#pragma unroll
            for (int c = 0; c < 16; ++c) a[k][c] += m * xv[c];
        }
    }
#pragma unroll
    for (int k = 0; k < 4; ++k) {
#pragma unroll
        for (int c = 0; c < 16; ++c) atomicAdd(&acc[(b * 4 + k) * 17 + c], a[k][c]);
        atomicAdd(&acc[(b * 4 + k) * 17 + 16], ms[k]);
    }
}

// proto -> l2-normalized prototypes pro (32,16)
__global__ void proto_norm_kernel(const float* __restrict__ acc, float* __restrict__ pro) {
    int i = threadIdx.x;
    if (i >= 32) return;
    float inv = 1.f / (acc[i * 17 + 16] + 1e-5f);
    float v[16]; float n2 = 0.f;
#pragma unroll
    for (int c = 0; c < 16; ++c) { v[c] = acc[i * 17 + c] * inv; n2 += v[c] * v[c]; }
    float nr = fmaxf(sqrtf(n2), 1e-12f);
#pragma unroll
    for (int c = 0; c < 16; ++c) pro[i * 16 + c] = v[c] / nr;
}

// ---------------------------------------------------------------------------
// Per-pixel similarity tail: self_simi, other (softmax over other batches),
// and entropy weight. sim (32 values) lives entirely in registers. xf NHWC.
// ---------------------------------------------------------------------------
__global__ void sim_final_kernel(const float* __restrict__ xf, const float* __restrict__ pro,
                                 float* __restrict__ selfS, float* __restrict__ otherS,
                                 float* __restrict__ wOut, int HW) {
    __shared__ float P[512];
    for (int i = threadIdx.x; i < 512; i += blockDim.x) P[i] = pro[i];
    __syncthreads();
    int idx = blockIdx.x * blockDim.x + threadIdx.x;
    int total = BN_B * HW;
    if (idx >= total) return;
    int b = idx / HW;
    float f[16]; float n2 = 0.f;
    const float* xv = xf + (size_t)idx * 16;   // contiguous feature
#pragma unroll
    for (int c = 0; c < 16; ++c) { f[c] = xv[c]; n2 += f[c] * f[c]; }
    float inv = 1.f / fmaxf(sqrtf(n2), 1e-12f);
#pragma unroll
    for (int c = 0; c < 16; ++c) f[c] *= inv;
    float sim[32];
#pragma unroll
    for (int j = 0; j < 32; ++j) {
        float s = 0.f;
#pragma unroll
        for (int c = 0; c < 16; ++c) s += f[c] * P[j * 16 + c];
        sim[j] = s;
    }
#pragma unroll
    for (int k = 0; k < 4; ++k) selfS[(size_t)idx * 4 + k] = sim[b * 4 + k];

    int counts[4] = {0, 0, 0, 0};
    for (int b2 = 0; b2 < BN_B; ++b2) {
        int arg = 0; float best = sim[b2 * 4];
        for (int k = 1; k < 4; ++k) {
            float v = sim[b2 * 4 + k];
            if (v > best) { best = v; arg = k; }  // first-max = jnp.argmax
        }
        counts[arg]++;
    }
    float ent = 0.f;
    for (int k2 = 0; k2 < 4; ++k2) {
        float a = (float)counts[k2] * 0.125f;
        ent -= a * logf(a + 1e-6f);
    }
    wOut[idx] = 1.f - ent / logf(4.0f);

    float ss[4] = {0, 0, 0, 0};
    for (int b2 = 0; b2 < BN_B; ++b2) {
        if (b2 == b) continue;
        for (int k = 0; k < 4; ++k) ss[k] += expf(sim[b2 * 4 + k]);
    }
    float itot = 1.f / (ss[0] + ss[1] + ss[2] + ss[3]);
    for (int k = 0; k < 4; ++k) otherS[(size_t)idx * 4 + k] = ss[k] * itot;
}

// ---------------------------------------------------------------------------
// Host-side helpers
// ---------------------------------------------------------------------------
struct UpP {
    const float *wp, *bp, *w1, *b1, *g1, *bt1, *w2, *b2, *g2, *bt2;
};

static int ilog2(int v) { int l = 0; while ((1 << l) < v) ++l; return l; }

static void launch_conv(hipStream_t stream, const _Float16* X, const _Float16* Apack,
                        const float* bias, float* Y, int Cin, int H, int W, int Cout,
                        bool outNCHW) {
    const int Npix = BN_B * H * W;
    const int tiles16 = (Cout + 15) / 16;
    const int cl2 = ilog2(Cin);
    dim3 blk(128);
    if (outNCHW) {
        dim3 g((Npix + 63) / 64, 1);
        conv3x3_wmma_kernel<1, true><<<g, blk, 0, stream>>>(X, Apack, bias, Y, BN_B, Cin, cl2, H, W, Cout);
    } else if (tiles16 >= 4) {
        dim3 g((Npix + 63) / 64, tiles16 / 4);
        conv3x3_wmma_kernel<4, false><<<g, blk, 0, stream>>>(X, Apack, bias, Y, BN_B, Cin, cl2, H, W, Cout);
    } else if (tiles16 >= 2) {
        dim3 g((Npix + 63) / 64, tiles16 / 2);
        conv3x3_wmma_kernel<2, false><<<g, blk, 0, stream>>>(X, Apack, bias, Y, BN_B, Cin, cl2, H, W, Cout);
    } else {
        dim3 g((Npix + 63) / 64, 1);
        conv3x3_wmma_kernel<1, false><<<g, blk, 0, stream>>>(X, Apack, bias, Y, BN_B, Cin, cl2, H, W, Cout);
    }
}

static int apack_halves(int Cout, int Cin) {
    int K = Cin * 9;
    return ((Cout + 15) / 16) * ((K + 31) / 32) * 512;
}

static void run_upblock(hipStream_t stream,
                        const _Float16* actPrev, int Hp, int Wp_, int CinPrev,
                        const float* xs, int Cs,
                        const UpP& u, const _Float16* w1p, const _Float16* w2p,
                        int Cmid, int Cout,
                        float* scProj, _Float16* scCat, float* scRaw, _Float16* scAct1,
                        float* scStats, _Float16* act2h, float* act2f) {
    const int Bn = BN_B;
    const int H2 = Hp * 2, W2 = Wp_ * 2;
    const int HWp = Hp * Wp_, HW2 = H2 * W2;
    const int Ccat = Cs + Cmid;
    const int Npx  = Bn * HW2;
    float* mean = scStats + 256;
    float* istd = scStats + 384;

    {   // 1x1 projection (NHWC in -> NCHW out for bilinear reads)
        int total = Bn * Cmid * HWp;
        conv1x1_kernel<<<(total + 255) / 256, 256, 0, stream>>>(
            actPrev, u.wp, u.bp, scProj, Bn, CinPrev, HWp, Cmid);
    }
    {   // bilinear 2x + concat -> NHWC f16 (one thread per pixel)
        upcat_nhwc_kernel<<<(Npx + 255) / 256, 256, 0, stream>>>(
            xs, scProj, scCat, Bn, Cs, Cmid, H2, W2);
    }
    {   // conv1 + BN + lrelu
        launch_conv(stream, scCat, w1p, u.b1, scRaw, Ccat, H2, W2, Cout, false);
        zero_f32_kernel<<<1, 256, 0, stream>>>(scStats, 256);
        bn_partial_kernel<<<64, 256, 0, stream>>>(scRaw, scStats, Cout, Npx);
        bn_finalize_kernel<<<1, 128, 0, stream>>>(scStats, mean, istd, Cout, (float)Npx);
        int total = Npx * Cout;
        bn_lrelu_kernel<<<(total + 255) / 256, 256, 0, stream>>>(
            scRaw, mean, istd, u.g1, u.bt1, scAct1, (float*)nullptr, Cout, total);
    }
    {   // conv2 + BN + lrelu
        launch_conv(stream, scAct1, w2p, u.b2, scRaw, Cout, H2, W2, Cout, false);
        zero_f32_kernel<<<1, 256, 0, stream>>>(scStats, 256);
        bn_partial_kernel<<<64, 256, 0, stream>>>(scRaw, scStats, Cout, Npx);
        bn_finalize_kernel<<<1, 128, 0, stream>>>(scStats, mean, istd, Cout, (float)Npx);
        int total = Npx * Cout;
        bn_lrelu_kernel<<<(total + 255) / 256, 256, 0, stream>>>(
            scRaw, mean, istd, u.g2, u.bt2, act2h, act2f, Cout, total);
    }
}

extern "C" void kernel_launch(void* const* d_in, const int* in_sizes, int n_in,
                              void* d_out, int out_size, void* d_ws, size_t ws_size,
                              hipStream_t stream) {
    (void)n_in; (void)out_size; (void)ws_size;
    // ------------------------------------------------------------------
    // Resolve input-index mapping (insertion-order vs sorted-key flatten)
    // ------------------------------------------------------------------
    int xi[5];
    int upBase[4];
    int S_b1 = 0, S_b2 = 1, S_bp = 2, S_bt1 = 3, S_bt2 = 4, S_g1 = 5, S_g2 = 6, S_w1 = 7, S_w2 = 8, S_wp = 9;
    int I_wp = 0, I_bp = 1, I_w1 = 2, I_b1 = 3, I_g1 = 4, I_bt1 = 5, I_w2 = 6, I_b2 = 7, I_g2 = 8, I_bt2 = 9;
    int owi, obi;
    bool sortedUp;
    if (in_sizes[0] == 4) {                 // whole tree sorted: ob,ow,up1..up4,x0..x4
        obi = 0; owi = 1;
        for (int u = 0; u < 4; ++u) upBase[u] = 2 + u * 10;
        for (int i = 0; i < 5; ++i) xi[i] = 42 + i;
        sortedUp = true;
    } else if (in_sizes[5] == 32768) {      // pure insertion order
        for (int i = 0; i < 5; ++i) xi[i] = i;
        for (int u = 0; u < 4; ++u) upBase[u] = 5 + u * 10;
        owi = 45; obi = 46;
        sortedUp = false;
    } else {                                // x0..x4 then sorted params
        for (int i = 0; i < 5; ++i) xi[i] = i;
        obi = 5; owi = 6;
        for (int u = 0; u < 4; ++u) upBase[u] = 7 + u * 10;
        sortedUp = true;
    }
    const float* x[5];
    for (int i = 0; i < 5; ++i) x[i] = (const float*)d_in[xi[i]];
    UpP U[4];
    for (int u = 0; u < 4; ++u) {
        int base = upBase[u];
        if (sortedUp) {
            U[u].wp  = (const float*)d_in[base + S_wp];  U[u].bp  = (const float*)d_in[base + S_bp];
            U[u].w1  = (const float*)d_in[base + S_w1];  U[u].b1  = (const float*)d_in[base + S_b1];
            U[u].g1  = (const float*)d_in[base + S_g1];  U[u].bt1 = (const float*)d_in[base + S_bt1];
            U[u].w2  = (const float*)d_in[base + S_w2];  U[u].b2  = (const float*)d_in[base + S_b2];
            U[u].g2  = (const float*)d_in[base + S_g2];  U[u].bt2 = (const float*)d_in[base + S_bt2];
        } else {
            U[u].wp  = (const float*)d_in[base + I_wp];  U[u].bp  = (const float*)d_in[base + I_bp];
            U[u].w1  = (const float*)d_in[base + I_w1];  U[u].b1  = (const float*)d_in[base + I_b1];
            U[u].g1  = (const float*)d_in[base + I_g1];  U[u].bt1 = (const float*)d_in[base + I_bt1];
            U[u].w2  = (const float*)d_in[base + I_w2];  U[u].b2  = (const float*)d_in[base + I_b2];
            U[u].g2  = (const float*)d_in[base + I_g2];  U[u].bt2 = (const float*)d_in[base + I_bt2];
        }
    }
    const float* ow = (const float*)d_in[owi];
    const float* ob = (const float*)d_in[obi];

    // ------------------------------------------------------------------
    // Workspace carve (bump allocator, 256B aligned)
    // ------------------------------------------------------------------
    size_t off = 0;
    char* base = (char*)d_ws;
    auto alloc = [&](size_t bytes) -> char* {
        char* p = base + off;
        off += (bytes + 255) & ~(size_t)255;
        return p;
    };
    const int convCout[4] = {128, 64, 32, 16};
    const int conv1Cin[4] = {256, 128, 64, 32};
    const int act2n[4] = {8 * 128 * 32 * 32, 8 * 64 * 64 * 64, 8 * 32 * 128 * 128, 8 * 16 * 256 * 256};

    _Float16* w1p[4]; _Float16* w2p[4]; _Float16* act2h[4];
    for (int u = 0; u < 4; ++u) w1p[u] = (_Float16*)alloc((size_t)apack_halves(convCout[u], conv1Cin[u]) * 2);
    for (int u = 0; u < 4; ++u) w2p[u] = (_Float16*)alloc((size_t)apack_halves(convCout[u], convCout[u]) * 2);
    _Float16* owp = (_Float16*)alloc((size_t)apack_halves(4, 16) * 2);
    _Float16* x4h = (_Float16*)alloc((size_t)524288 * 2);
    for (int u = 0; u < 4; ++u) act2h[u] = (_Float16*)alloc((size_t)act2n[u] * 2);
    float*     xf32    = (float*)alloc((size_t)8388608 * 4);
    float*     scProj  = (float*)alloc((size_t)2097152 * 4);
    _Float16*  scCat   = (_Float16*)alloc((size_t)16777216 * 2);
    float*     scRaw   = (float*)alloc((size_t)8388608 * 4);
    _Float16*  scAct1  = (_Float16*)alloc((size_t)8388608 * 2);
    float*     scStats = (float*)alloc(512 * 4);
    float*     protoAcc = (float*)alloc(544 * 4);
    float*     proN     = (float*)alloc(512 * 4);

    // ------------------------------------------------------------------
    // Pack conv weights into WMMA A-fragment layout (tap-major, f16)
    // ------------------------------------------------------------------
    for (int u = 0; u < 4; ++u) {
        int n1 = apack_halves(convCout[u], conv1Cin[u]);
        pack_wgt_kernel<<<(n1 + 255) / 256, 256, 0, stream>>>(U[u].w1, w1p[u], convCout[u], conv1Cin[u]);
        int n2 = apack_halves(convCout[u], convCout[u]);
        pack_wgt_kernel<<<(n2 + 255) / 256, 256, 0, stream>>>(U[u].w2, w2p[u], convCout[u], convCout[u]);
    }
    {
        int n3 = apack_halves(4, 16);
        pack_wgt_kernel<<<(n3 + 255) / 256, 256, 0, stream>>>(ow, owp, 4, 16);
    }
    // x4 NCHW f32 -> NHWC f16
    nchw_to_nhwc_f16_kernel<<<(BN_B * 256 + 255) / 256, 256, 0, stream>>>(x[4], x4h, BN_B, 256, 256);

    // ------------------------------------------------------------------
    // Decoder stages
    // ------------------------------------------------------------------
    run_upblock(stream, x4h,      16,  16,  256, x[3], 128, U[0], w1p[0], w2p[0], 128, 128,
                scProj, scCat, scRaw, scAct1, scStats, act2h[0], nullptr);
    run_upblock(stream, act2h[0], 32,  32,  128, x[2], 64,  U[1], w1p[1], w2p[1], 64,  64,
                scProj, scCat, scRaw, scAct1, scStats, act2h[1], nullptr);
    run_upblock(stream, act2h[1], 64,  64,  64,  x[1], 32,  U[2], w1p[2], w2p[2], 32,  32,
                scProj, scCat, scRaw, scAct1, scStats, act2h[2], nullptr);
    run_upblock(stream, act2h[2], 128, 128, 32,  x[0], 16,  U[3], w1p[3], w2p[3], 16,  16,
                scProj, scCat, scRaw, scAct1, scStats, act2h[3], xf32);

    // ------------------------------------------------------------------
    // Final 3x3 conv -> logits in d_out[0 : 8*4*256*256), NCHW
    // ------------------------------------------------------------------
    float* outLogits = (float*)d_out;
    launch_conv(stream, act2h[3], owp, ob, outLogits, 16, 256, 256, 4, true);

    // ------------------------------------------------------------------
    // Prototypes + similarity tail
    // ------------------------------------------------------------------
    zero_f32_kernel<<<3, 256, 0, stream>>>(protoAcc, 544);
    proto_accum_kernel<<<8, 256, 0, stream>>>(outLogits, xf32, protoAcc, 65536);
    proto_norm_kernel<<<1, 32, 0, stream>>>(protoAcc, proN);

    float* selfS  = outLogits + 2097152;
    float* otherS = outLogits + 2 * 2097152;
    float* wOut   = outLogits + 3 * 2097152;
    {
        int total = 8 * 65536;
        sim_final_kernel<<<(total + 255) / 256, 256, 0, stream>>>(xf32, proN, selfS, otherS, wOut, 65536);
    }
}
